// TorchGatedDeltaNetSequenceMixer_47064251630315
// MI455X (gfx1250) — compile-verified, matrix operand order checked by reading, not run on hardware
//
#include <hip/hip_runtime.h>
#include <hip/hip_bf16.h>
#include <stdint.h>

typedef __attribute__((ext_vector_type(16))) _Float16 v16h;
typedef __attribute__((ext_vector_type(8)))  float    v8f;

#define BB    4
#define SS    2048
#define DD    1024
#define HH    16
#define HDD   64
#define KW    4
#define ROWS  (BB*SS)          // 8192
#define NQKV  3072
#define NCTRL 1056
#define NCTRLP 1088            // padded to multiple of 64 for tiling
#define EPSF  1e-6f
#define CHUNK 64
#define NCHUNK (SS/CHUNK)      // 32

#define TM 64
#define TN 64
#define TK 32

__device__ __forceinline__ float sigmoidf_(float x) { return 1.0f / (1.0f + __expf(-x)); }
__device__ __forceinline__ float siluf_(float x)    { return x / (1.0f + __expf(-x)); }

union Frag { v16h v; uint32_t u[8]; };

// per-lane 16x32 f16 fragment: two contiguous 16B runs at +sel8 and +16+sel8
__device__ __forceinline__ Frag ldfrag_(const _Float16* base) {
    Frag f;
    const uint32_t* p0 = (const uint32_t*)base;
#pragma unroll
    for (int i = 0; i < 4; ++i) f.u[i] = p0[i];
    const uint32_t* p1 = (const uint32_t*)(base + 16);
#pragma unroll
    for (int i = 0; i < 4; ++i) f.u[4 + i] = p1[i];
    return f;
}

// two 16x16 tiles of C[64x64] += A[64x64] @ W[64x64]^T, operands f16 in LDS
// (row stride 64). Wave tile: rows 16*mi.., cols nh*32.. and nh*32+16..
__device__ __forceinline__ void gemm64_(const _Float16* As, const _Float16* Ws,
                                        int mi, int nh, int lane,
                                        v8f& acc0, v8f& acc1) {
    const int sel8 = (lane >> 4) << 3;
    const int ar  = mi * 16 + (lane & 15);
    const int br0 = nh * 32 + (lane & 15);
    const int br1 = br0 + 16;
#pragma unroll
    for (int kc = 0; kc < 64; kc += 32) {
        Frag a  = ldfrag_(As + ar * 64 + kc + sel8);
        Frag b0 = ldfrag_(Ws + br0 * 64 + kc + sel8);
        Frag b1 = ldfrag_(Ws + br1 * 64 + kc + sel8);
        acc0 = __builtin_amdgcn_wmma_f32_16x16x32_f16(
                   false, a.v, false, b0.v, (short)0, acc0, false, false);
        acc1 = __builtin_amdgcn_wmma_f32_16x16x32_f16(
                   false, a.v, false, b1.v, (short)0, acc1, false, false);
    }
}

// ---------------------------------------------------------------- casts
__global__ void cast_f32_f16_kernel(const float* __restrict__ src,
                                    _Float16* __restrict__ dst, size_t n) {
    size_t i = (size_t)blockIdx.x * blockDim.x + threadIdx.x;
    if (i < n) dst[i] = (_Float16)src[i];
}

__global__ void cast_pad_f32_f16_kernel(const float* __restrict__ src,
                                        _Float16* __restrict__ dst,
                                        size_t nsrc, size_t ndst) {
    size_t i = (size_t)blockIdx.x * blockDim.x + threadIdx.x;
    if (i < ndst) dst[i] = (i < nsrc) ? (_Float16)src[i] : (_Float16)0.0f;
}

// ---------------------------------------------------------------- WMMA GEMM
// C[M,N] = A[M,K] @ W[N,K]^T, 64x64 block tile, async-LDS double buffering.
template <bool HALF_OUT>
__global__ __launch_bounds__(256)
void wmma_gemm_lds_kernel(const _Float16* __restrict__ A,
                          const _Float16* __restrict__ W,
                          const float* __restrict__ bias, int biasN,
                          void* __restrict__ Cout,
                          int M, int N, int K) {
    __shared__ __align__(16) _Float16 Atile[2][TM * TK];
    __shared__ __align__(16) _Float16 Btile[2][TN * TK];

    const int tid  = threadIdx.x;
    const int lane = tid & 31;
    const int wv   = tid >> 5;
    const int mi   = wv & 3;
    const int nh   = wv >> 2;

    const int mrow0 = blockIdx.y * TM;
    const int ncol0 = blockIdx.x * TN;

    const int arow = tid >> 2;
    const int achk = (tid & 3) * 8;
    const _Float16* Ag = A + (size_t)(mrow0 + arow) * K + achk;
    const _Float16* Wg = W + (size_t)(ncol0 + arow) * K + achk;

    auto issue = [&](int buf, int kc) {
        uint32_t la = (uint32_t)(uintptr_t)(&Atile[buf][arow * TK + achk]);
        uint32_t lb = (uint32_t)(uintptr_t)(&Btile[buf][arow * TK + achk]);
        const _Float16* ga = Ag + kc;
        const _Float16* gb = Wg + kc;
        asm volatile("global_load_async_to_lds_b128 %0, %1, off"
                     :: "v"(la), "v"(ga) : "memory");
        asm volatile("global_load_async_to_lds_b128 %0, %1, off"
                     :: "v"(lb), "v"(gb) : "memory");
    };

    const int sel8 = (lane >> 4) << 3;
    const int afr  = mi * 16 + (lane & 15);
    const int bfr0 = nh * 32 + (lane & 15);
    const int bfr1 = bfr0 + 16;

    v8f acc0 = {}, acc1 = {};
    issue(0, 0);

    for (int kc = 0; kc < K; kc += TK) {
        const int  buf  = (kc / TK) & 1;
        const bool more = (kc + TK) < K;
        if (more) {
            issue(buf ^ 1, kc + TK);
            asm volatile("s_wait_asynccnt 2" ::: "memory");
        } else {
            asm volatile("s_wait_asynccnt 0" ::: "memory");
        }
        __syncthreads();

        Frag a  = ldfrag_(&Atile[buf][afr * TK + sel8]);
        Frag b0 = ldfrag_(&Btile[buf][bfr0 * TK + sel8]);
        Frag b1 = ldfrag_(&Btile[buf][bfr1 * TK + sel8]);

        acc0 = __builtin_amdgcn_wmma_f32_16x16x32_f16(
                   false, a.v, false, b0.v, (short)0, acc0, false, false);
        acc1 = __builtin_amdgcn_wmma_f32_16x16x32_f16(
                   false, a.v, false, b1.v, (short)0, acc1, false, false);
        __syncthreads();
    }

    const int rsel  = (lane >> 4) << 3;
    const int col0w = ncol0 + nh * 32 + (lane & 15);
    const int col1w = col0w + 16;
#pragma unroll
    for (int r = 0; r < 8; ++r) {
        const int row = mrow0 + mi * 16 + r + rsel;
        float v0 = acc0[r];
        float v1 = acc1[r];
        if (bias) {
            if (col0w < biasN) v0 += bias[col0w];
            if (col1w < biasN) v1 += bias[col1w];
        }
        if (HALF_OUT) {
            ((_Float16*)Cout)[(size_t)row * N + col0w] = (_Float16)v0;
            ((_Float16*)Cout)[(size_t)row * N + col1w] = (_Float16)v1;
        } else {
            ((float*)Cout)[(size_t)row * N + col0w] = v0;
            ((float*)Cout)[(size_t)row * N + col1w] = v1;
        }
    }
}

// ---------------------------------------------------------------- conv + gates
__global__ __launch_bounds__(1024)
void conv_gate_kernel(const _Float16* __restrict__ qkv,
                      const float* __restrict__ ctrl,
                      const float* __restrict__ wq,
                      const float* __restrict__ wk,
                      const float* __restrict__ wv,
                      float* __restrict__ qbuf,
                      float* __restrict__ kbuf,
                      float* __restrict__ vbuf,
                      float* __restrict__ alpha,
                      float* __restrict__ beta) {
    const int pos = blockIdx.x;
    const int b   = pos / SS;
    const int s   = pos - b * SS;
    const int d   = threadIdx.x;
    const int h   = d >> 6;
    const int hd  = d & 63;
    const int lane = d & 31;
    const int wvid = d >> 5;

    float yq = 0.f, yk = 0.f, yv = 0.f;
#pragma unroll
    for (int j = 0; j < KW; ++j) {
        const int sj = s - (KW - 1) + j;
        if (sj >= 0) {
            const size_t base = (size_t)(b * SS + sj) * NQKV;
            const float xq = (float)qkv[base + d];
            const float xk = (float)qkv[base + DD + d];
            const float xv = (float)qkv[base + 2 * DD + d];
            yq = fmaf(xq, wq[d * KW + j], yq);
            yk = fmaf(xk, wk[d * KW + j], yk);
            yv = fmaf(xv, wv[d * KW + j], yv);
        }
    }
    yq = siluf_(yq); yk = siluf_(yk); yv = siluf_(yv);

    __shared__ float pq[32], pk[32];
    float sq = yq * yq, sk = yk * yk;
#pragma unroll
    for (int off = 16; off; off >>= 1) {
        sq += __shfl_xor(sq, off, 32);
        sk += __shfl_xor(sk, off, 32);
    }
    if (lane == 0) { pq[wvid] = sq; pk[wvid] = sk; }
    __syncthreads();
    const float nq = sqrtf(pq[2 * h] + pq[2 * h + 1]);
    const float nk = sqrtf(pk[2 * h] + pk[2 * h + 1]);
    const float qn = yq / fmaxf(nq, EPSF);
    const float kn = yk / fmaxf(nk, EPSF);

    const size_t obase = ((size_t)(b * HH + h) * SS + s) * HDD + hd;
    qbuf[obase] = qn;
    kbuf[obase] = kn;
    vbuf[obase] = yv;

    if (hd == 0) {
        const size_t crow = (size_t)pos * NCTRLP;
        alpha[(size_t)(b * HH + h) * SS + s] = sigmoidf_(ctrl[crow + h]) * 0.98f + 0.01f;
        beta [(size_t)(b * HH + h) * SS + s] = sigmoidf_(ctrl[crow + HH + h]);
    }
}

// ---------------------------------------------------------------- chunked delta-rule scan (WY form, all-WMMA)
// One block per (b,h); 256 threads / 8 waves; 32 serial chunks of 64 steps.
// State S (64x64 f32) persists in WMMA accumulators across chunks.
__global__ __launch_bounds__(256)
void chunk_scan_kernel(const float* __restrict__ qbuf,
                       const float* __restrict__ kbuf,
                       const float* __restrict__ vbuf,
                       const float* __restrict__ alpha,
                       const float* __restrict__ beta,
                       const float* __restrict__ norm_w,
                       _Float16* __restrict__ normed) {     // (ROWS, 1024)
    const int bh   = blockIdx.x;
    const int b    = bh >> 4;
    const int h    = bh & 15;
    const int tid  = threadIdx.x;
    const int lane = tid & 31;
    const int wv   = tid >> 5;
    const int mi   = wv & 3;
    const int nh   = wv >> 2;
    const int rsel = (lane >> 4) << 3;

    const float* Qg = qbuf + (size_t)bh * SS * HDD;
    const float* Kg = kbuf + (size_t)bh * SS * HDD;
    const float* Vg = vbuf + (size_t)bh * SS * HDD;
    const float* Ab = alpha + (size_t)bh * SS;
    const float* Bb = beta  + (size_t)bh * SS;

    // -------- overlaid LDS layout (56 KB) --------
    __shared__ __align__(16) char smem[57344];
    _Float16* Kh   = (_Float16*)(smem);            // [0,8K)   stage..2a
    _Float16* Qh   = (_Float16*)(smem + 8192);     // [8K,16K) stage..2a
    float*    Amat = (float*)   (smem);            // [0,16K)  2b..3 (overlay)
    float*    Obuf = (float*)   (smem);            // [0,16K)  4..5  (overlay)
    _Float16* KTs  = (_Float16*)(smem + 16384);    // s-scaled K^T, stage..4
    _Float16* SH   = (_Float16*)(smem + 24576);    // S as f16, 4..next 2a
    _Float16* rhs16= (_Float16*)(smem + 32768);    // 2b..3
    _Float16* Lh   = (_Float16*)(smem + 40960);    // 2b..4
    _Float16* RT   = (_Float16*)(smem + 49152);    // R^T, 3..4
    __shared__ float Gc[64], Bt[64], sArr[64], normw_s[64];
    __shared__ float wtot[2];

    // prologue: zero state, cache norm weights
    for (int i = tid; i < 4096; i += 256) SH[i] = (_Float16)0.f;
    if (tid < 64) normw_s[tid] = norm_w[tid];
    v8f accS0 = {}, accS1 = {};

    for (int chunk = 0; chunk < NCHUNK; ++chunk) {
        const int s0 = chunk * CHUNK;

        // ---- phase 1: log-decay prefix + staging ----
        float pg = 0.f;
        if (tid < 64) {
            Bt[tid] = Bb[s0 + tid];
            pg = __logf(Ab[s0 + tid]);
#pragma unroll
            for (int o2 = 1; o2 < 32; o2 <<= 1) {
                float nx = __shfl_up(pg, o2, 32);
                if (lane >= o2) pg += nx;
            }
            if (lane == 31) wtot[wv] = pg;
        }
        __syncthreads();
        if (tid < 64) {
            if (wv == 1) pg += wtot[0];
            Gc[tid] = pg;
        }
        __syncthreads();
        if (tid < 64) sArr[tid] = Bt[tid] * __expf(Gc[63] - Gc[tid]);
        __syncthreads();

        for (int i = tid; i < 4096; i += 256) {
            const int rr = i >> 6, cc = i & 63;
            const float kv = Kg[(size_t)s0 * 64 + i];
            Kh[i] = (_Float16)kv;
            KTs[cc * 64 + rr] = (_Float16)(sArr[rr] * kv);
            Qh[i] = (_Float16)Qg[(size_t)s0 * 64 + i];
        }
        __syncthreads();

        // ---- phase 2a: four 64^3 WMMA GEMMs into registers ----
        v8f kk0 = {}, kk1 = {}, qk0 = {}, qk1 = {};
        v8f zk0 = {}, zk1 = {}, zq0 = {}, zq1 = {};
        gemm64_(Kh, Kh, mi, nh, lane, kk0, kk1);   // K K^T
        gemm64_(Qh, Kh, mi, nh, lane, qk0, qk1);   // Q K^T
        gemm64_(Kh, SH, mi, nh, lane, zk0, zk1);   // K S0^T
        gemm64_(Qh, SH, mi, nh, lane, zq0, zq1);   // Q S0^T
        __syncthreads();   // Kh/Qh region free -> Amat

        // ---- phase 2b: decay/beta scaling epilogues ----
#pragma unroll
        for (int r = 0; r < 8; ++r) {
            const int row = mi * 16 + r + rsel;        // t
            const int c0  = nh * 32 + (lane & 15);     // j (or v)
            const int c1  = c0 + 16;
            const float gt  = Gc[row];
            const float egt = __expf(gt);
            const float w0 = Bt[c0] * __expf(gt - Gc[c0]);
            const float w1 = Bt[c1] * __expf(gt - Gc[c1]);
            Amat[row * 64 + c0] = (c0 < row) ? kk0[r] * w0 : 0.f;
            Amat[row * 64 + c1] = (c1 < row) ? kk1[r] * w1 : 0.f;
            Lh[row * 64 + c0] = (c0 <= row) ? (_Float16)(qk0[r] * w0) : (_Float16)0.f;
            Lh[row * 64 + c1] = (c1 <= row) ? (_Float16)(qk1[r] * w1) : (_Float16)0.f;
            const float vv0 = Vg[(size_t)(s0 + row) * 64 + c0];
            const float vv1 = Vg[(size_t)(s0 + row) * 64 + c1];
            rhs16[row * 64 + c0] = (_Float16)(vv0 - egt * zk0[r]);
            rhs16[row * 64 + c1] = (_Float16)(vv1 - egt * zk1[r]);
            zq0[r] *= egt;
            zq1[r] *= egt;
        }
        __syncthreads();

        // ---- phase 3: forward substitution (I+A) R = rhs, column-parallel ----
        if (tid < 64) {
            float R[64];
#pragma unroll
            for (int t = 0; t < 64; ++t) {
                float acc = (float)rhs16[t * 64 + tid];
#pragma unroll
                for (int j = 0; j < t; ++j) acc -= Amat[t * 64 + j] * R[j];
                R[t] = acc;
            }
#pragma unroll
            for (int t = 0; t < 64; ++t) RT[tid * 64 + t] = (_Float16)R[t];
        }
        __syncthreads();   // Amat dead -> Obuf

        // ---- phase 4: O = Gamma.QS + L R ; S = e^G S + (sR)^T K ----
        gemm64_(Lh, RT, mi, nh, lane, zq0, zq1);
        const float eC = __expf(Gc[63]);
#pragma unroll
        for (int r = 0; r < 8; ++r) { accS0[r] *= eC; accS1[r] *= eC; }
        gemm64_(RT, KTs, mi, nh, lane, accS0, accS1);
#pragma unroll
        for (int r = 0; r < 8; ++r) {
            const int row = mi * 16 + r + rsel;
            const int c0  = nh * 32 + (lane & 15);
            const int c1  = c0 + 16;
            Obuf[row * 64 + c0] = zq0[r];
            Obuf[row * 64 + c1] = zq1[r];
            SH[row * 64 + c0] = (_Float16)accS0[r];
            SH[row * 64 + c1] = (_Float16)accS1[r];
        }
        __syncthreads();

        // ---- phase 5: fused RMSNorm + store ----
        if (tid < 64) {
            const float4* orow = (const float4*)(Obuf + tid * 64);
            float ssum = 0.f;
#pragma unroll
            for (int j4 = 0; j4 < 16; ++j4) {
                const float4 o4 = orow[j4];
                ssum += o4.x * o4.x + o4.y * o4.y + o4.z * o4.z + o4.w * o4.w;
            }
            const float rms = rsqrtf(ssum * (1.f / 64.f) + EPSF);
            const size_t gb = (size_t)(b * SS + s0 + tid) * DD + h * 64;
#pragma unroll
            for (int j4 = 0; j4 < 16; ++j4) {
                const float4 o4 = orow[j4];
                normed[gb + 4 * j4 + 0] = (_Float16)(o4.x * rms * normw_s[4 * j4 + 0]);
                normed[gb + 4 * j4 + 1] = (_Float16)(o4.y * rms * normw_s[4 * j4 + 1]);
                normed[gb + 4 * j4 + 2] = (_Float16)(o4.z * rms * normw_s[4 * j4 + 2]);
                normed[gb + 4 * j4 + 3] = (_Float16)(o4.w * rms * normw_s[4 * j4 + 3]);
            }
        }
        __syncthreads();
    }
}

// ---------------------------------------------------------------- gated epilogue
__global__ void epilogue_kernel(const float* __restrict__ ctrl,
                                const float* __restrict__ out_b,
                                float* __restrict__ out) {
    size_t i = (size_t)blockIdx.x * blockDim.x + threadIdx.x;
    if (i >= (size_t)ROWS * DD) return;
    const int row = (int)(i >> 10);
    const int d   = (int)(i & 1023);
    const float g = sigmoidf_(ctrl[(size_t)row * NCTRLP + 2 * HH + d]);
    out[i] = g * (out[i] + out_b[d]);
}

// ================================================================ launch
extern "C" void kernel_launch(void* const* d_in, const int* in_sizes, int n_in,
                              void* d_out, int out_size, void* d_ws, size_t ws_size,
                              hipStream_t stream) {
    const float* hidden = (const float*)d_in[0];
    const float* qkv_w  = (const float*)d_in[1];
    const float* ctrl_w = (const float*)d_in[2];
    const float* ctrl_b = (const float*)d_in[3];
    const float* convq  = (const float*)d_in[4];
    const float* convk  = (const float*)d_in[5];
    const float* convv  = (const float*)d_in[6];
    const float* norm_w = (const float*)d_in[7];
    const float* out_w  = (const float*)d_in[8];
    const float* out_b  = (const float*)d_in[9];
    float* out = (float*)d_out;

    char* ws = (char*)d_ws;
    size_t off = 0;
    auto alloc = [&](size_t bytes) -> char* {
        char* p = ws + off;
        off = (off + bytes + 255) & ~(size_t)255;
        return p;
    };

    _Float16* hidden_h = (_Float16*)alloc((size_t)ROWS * DD * 2);
    _Float16* qkvw_h   = (_Float16*)alloc((size_t)NQKV * DD * 2);
    _Float16* ctrlw_h  = (_Float16*)alloc((size_t)NCTRLP * DD * 2);
    _Float16* outw_h   = (_Float16*)alloc((size_t)DD * DD * 2);
    _Float16* qkv_h    = (_Float16*)alloc((size_t)ROWS * NQKV * 2);
    float*    ctrl     = (float*)   alloc((size_t)ROWS * NCTRLP * 4);
    float*    qbuf     = (float*)   alloc((size_t)ROWS * DD * 4);
    float*    kbuf     = (float*)   alloc((size_t)ROWS * DD * 4);
    float*    vbuf     = (float*)   alloc((size_t)ROWS * DD * 4);
    float*    alpha    = (float*)   alloc((size_t)BB * HH * SS * 4);
    float*    beta     = (float*)   alloc((size_t)BB * HH * SS * 4);
    _Float16* normed_h = (_Float16*)alloc((size_t)ROWS * DD * 2);

    {
        size_t n;
        n = (size_t)ROWS * DD;
        cast_f32_f16_kernel<<<(n + 255) / 256, 256, 0, stream>>>(hidden, hidden_h, n);
        n = (size_t)NQKV * DD;
        cast_f32_f16_kernel<<<(n + 255) / 256, 256, 0, stream>>>(qkv_w, qkvw_h, n);
        size_t ns = (size_t)NCTRL * DD, nd = (size_t)NCTRLP * DD;
        cast_pad_f32_f16_kernel<<<(nd + 255) / 256, 256, 0, stream>>>(ctrl_w, ctrlw_h, ns, nd);
        n = (size_t)DD * DD;
        cast_f32_f16_kernel<<<(n + 255) / 256, 256, 0, stream>>>(out_w, outw_h, n);
    }

    wmma_gemm_lds_kernel<true><<<dim3(NQKV / TN, ROWS / TM), 256, 0, stream>>>(
        hidden_h, qkvw_h, nullptr, 0, qkv_h, ROWS, NQKV, DD);

    wmma_gemm_lds_kernel<false><<<dim3(NCTRLP / TN, ROWS / TM), 256, 0, stream>>>(
        hidden_h, ctrlw_h, ctrl_b, NCTRL, ctrl, ROWS, NCTRLP, DD);

    conv_gate_kernel<<<ROWS, 1024, 0, stream>>>(
        qkv_h, ctrl, convq, convk, convv, qbuf, kbuf, vbuf, alpha, beta);

    chunk_scan_kernel<<<BB * HH, 256, 0, stream>>>(
        qbuf, kbuf, vbuf, alpha, beta, norm_w, normed_h);

    wmma_gemm_lds_kernel<false><<<dim3(DD / TN, ROWS / TM), 256, 0, stream>>>(
        normed_h, outw_h, nullptr, 0, out, ROWS, DD, DD);

    {
        size_t n = (size_t)ROWS * DD;
        epilogue_kernel<<<(n + 255) / 256, 256, 0, stream>>>(ctrl, out_b, out);
    }
}
